// RoPEMultiheadAttention_84112639525293
// MI455X (gfx1250) — matrix operand order repaired and checked by reading, not compiled, not run on hardware
//
#include <hip/hip_runtime.h>
#include <hip/hip_bf16.h>

#define B_ 4
#define S_ 2048
#define E_ 1024
#define H_ 16
#define D_ 64

typedef __attribute__((ext_vector_type(16))) __bf16 v16bf;
typedef __attribute__((ext_vector_type(8)))  __bf16 v8bf;
typedef __attribute__((ext_vector_type(2)))  __bf16 v2bf;
typedef __attribute__((ext_vector_type(8)))  float  v8f;
typedef __attribute__((ext_vector_type(2)))  float  v2f;
typedef __attribute__((ext_vector_type(8)))  unsigned short v8us;

// scalar f32 -> bf16 bits (RNE via compiler-legalized fptrunc)
__device__ __forceinline__ unsigned short f2bfbits(float f) {
    __bf16 h = (__bf16)f;
    unsigned short u;
    __builtin_memcpy(&u, &h, 2);
    return u;
}
// packed pair f32x2 -> bf16x2 (selects v_cvt_pk_bf16_f32 when available)
__device__ __forceinline__ unsigned f2bf2bits(float a, float b) {
    v2f in = {a, b};
    v2bf p = __builtin_convertvector(in, v2bf);
    unsigned u;
    __builtin_memcpy(&u, &p, 4);
    return u;
}
__device__ __forceinline__ float bf2f(unsigned short h) {
    unsigned u = (unsigned)h << 16;
    float f;
    __builtin_memcpy(&f, &u, 4);
    return f;
}
__device__ __forceinline__ unsigned long long pack4(float a, float b, float c, float d) {
    return (unsigned long long)f2bf2bits(a, b)
         | ((unsigned long long)f2bf2bits(c, d) << 32);
}
// build a 16-element bf16 fragment from two contiguous 16-byte chunks
__device__ __forceinline__ v16bf frag2(const unsigned short* p0, const unsigned short* p1) {
    v8bf lo, hi;
    __builtin_memcpy(&lo, p0, 16);
    __builtin_memcpy(&hi, p1, 16);
    return __builtin_shufflevector(lo, hi, 0,1,2,3,4,5,6,7,8,9,10,11,12,13,14,15);
}
#define WMMA_BF16(A, Bf, C) \
    __builtin_amdgcn_wmma_f32_16x16x32_bf16(false, (A), false, (Bf), (short)0, (C), false, false)

// ---------------------------------------------------------------------------
// Kernel 1: QKV projection (X @ W^T + b) + RoPE(Q,K).
// Q,K -> bf16 [B,H,S,D]; V -> bf16 [B,H,D,S] (transposed for P@V).
// block tile 128(M) x 64(N), 8 waves, wave: 16 rows x 64 cols (4 accumulators)
// A-frag lane: K = hi*8+e | 16+hi*8+e (contiguous 8+8).  B-frag: same.
// ---------------------------------------------------------------------------
__global__ void __launch_bounds__(256) qkv_rope_kernel(
    const float* __restrict__ q_in, const float* __restrict__ k_in,
    const float* __restrict__ v_in,
    const float* __restrict__ Wq, const float* __restrict__ bq,
    const float* __restrict__ Wk, const float* __restrict__ bk,
    const float* __restrict__ Wv, const float* __restrict__ bv,
    unsigned short* __restrict__ Qh, unsigned short* __restrict__ Kh,
    unsigned short* __restrict__ Vt)
{
    const int proj = blockIdx.z;
    const float* X    = (proj == 0) ? q_in : (proj == 1) ? k_in : v_in;
    const float* W    = (proj == 0) ? Wq   : (proj == 1) ? Wk   : Wv;
    const float* bias = (proj == 0) ? bq   : (proj == 1) ? bk   : bv;
    unsigned short* out = (proj == 0) ? Qh : (proj == 1) ? Kh : Vt;

    const int tileM = blockIdx.x * 128;
    const int head  = blockIdx.y;          // one 64-col N tile == one head
    const int tileN = head * 64;
    const int tid   = threadIdx.x;
    const int wave  = tid >> 5;
    const int lane  = tid & 31;
    const int hi    = (lane >> 4) & 1;
    const int col16 = lane & 15;
    const int rowBase = tileM + wave * 16;

    __shared__ unsigned short As[128][40];   // [M][K] bf16, 80B stride
    __shared__ unsigned short Bs[64][40];    // [N][K] bf16, 80B stride

    v8f acc[4] = {};

    for (int k0 = 0; k0 < E_; k0 += 32) {
        // stage A tile 128x32: float4 reads, packed-cvt 8B writes
        #pragma unroll
        for (int e = tid; e < 128 * 8; e += 256) {
            int r = e >> 3, c4 = (e & 7) * 4;
            float4 x = *reinterpret_cast<const float4*>(
                X + (size_t)(tileM + r) * E_ + k0 + c4);
            unsigned long long p = pack4(x.x, x.y, x.z, x.w);
            __builtin_memcpy(&As[r][c4], &p, 8);
        }
        // stage B tile: Bs[n][k] = W[tileN+n][k0+k]
        #pragma unroll
        for (int e = tid; e < 64 * 8; e += 256) {
            int n = e >> 3, c4 = (e & 7) * 4;
            float4 w = *reinterpret_cast<const float4*>(
                W + (size_t)(tileN + n) * E_ + k0 + c4);
            unsigned long long p = pack4(w.x, w.y, w.z, w.w);
            __builtin_memcpy(&Bs[n][c4], &p, 8);
        }
        if (k0 + 32 < E_)
            __builtin_prefetch(&X[(size_t)(tileM + wave * 16) * E_ + k0 + 32], 0, 1);
        __syncthreads();

        v16bf a = frag2(&As[wave * 16 + col16][hi * 8],
                        &As[wave * 16 + col16][16 + hi * 8]);
        #pragma unroll
        for (int j = 0; j < 4; ++j) {
            v16bf bf = frag2(&Bs[j * 16 + col16][hi * 8],
                             &Bs[j * 16 + col16][16 + hi * 8]);
            acc[j] = WMMA_BF16(a, bf, acc[j]);
        }
        __syncthreads();
    }

    // bias
    #pragma unroll
    for (int j = 0; j < 4; ++j) {
        float bb = bias[tileN + j * 16 + col16];
        #pragma unroll
        for (int i = 0; i < 8; ++i) acc[j][i] += bb;
    }

    // RoPE for Q,K: (d, d+32) pair lives in acc tiles (j, j+2), same lane/i
    if (proj < 2) {
        #pragma unroll
        for (int i = 0; i < 8; ++i) {
            int m = rowBase + i + hi * 8;
            float pos = (float)(m & (S_ - 1));
            #pragma unroll
            for (int j = 0; j < 2; ++j) {
                float dlo = (float)(j * 16 + col16);     // 0..31
                // inv_freq = 10000^(-2*dlo/64) = exp(-ln(1e4)/32 * dlo)
                float ang = pos * __expf(-0.28782313662425572f * dlo);
                float sn, cs;
                __sincosf(ang, &sn, &cs);
                float x = acc[j][i], y = acc[j + 2][i];
                acc[j][i]     = x * cs - y * sn;
                acc[j + 2][i] = y * cs + x * sn;
            }
        }
    }

    #pragma unroll
    for (int j = 0; j < 4; ++j) {
        int d = j * 16 + col16;
        #pragma unroll
        for (int i = 0; i < 8; ++i) {
            int m = rowBase + i + hi * 8;
            int b = m >> 11;
            int s = m & (S_ - 1);
            size_t idx = (proj == 2)
                ? ((((size_t)b * H_ + head) * D_ + d) * S_ + s)    // V: [B,H,D,S]
                : ((((size_t)b * H_ + head) * S_ + s) * D_ + d);   // Q,K: [B,H,S,D]
            out[idx] = f2bfbits(acc[j][i]);
        }
    }
}

// ---------------------------------------------------------------------------
// Kernel 2: attention. WG = (batch, 16 query rows), loops over 16 heads.
// LDS: pbf bf16[16][2048] (exp values) + avg f32[16][2048] + reductions.
// ---------------------------------------------------------------------------
__global__ void __launch_bounds__(256) attn_kernel(
    const unsigned short* __restrict__ Qh, const unsigned short* __restrict__ Kh,
    const unsigned short* __restrict__ Vt,
    unsigned short* __restrict__ AO, float* __restrict__ attn_avg)
{
    extern __shared__ float smem[];
    unsigned short* pbf = (unsigned short*)smem;      // 16*2048 bf16 (64KB)
    float* avg   = smem + 16384;                      // 16*2048 f32  (128KB)
    float* red   = avg + 16 * 2048;                   // 16*16
    float* invs  = red + 256;                         // 16
    float* pvred = invs + 16;                         // 4*16*16 split-K partials

    const int b     = blockIdx.y;
    const int q0    = blockIdx.x * 16;
    const int tid   = threadIdx.x;
    const int wave  = tid >> 5;
    const int lane  = tid & 31;
    const int hi    = (lane >> 4) & 1;
    const int col16 = lane & 15;

    for (int e = tid * 4; e < 16 * 2048; e += 1024)
        *reinterpret_cast<float4*>(avg + e) = make_float4(0.f, 0.f, 0.f, 0.f);
    __syncthreads();

    for (int h = 0; h < H_; ++h) {
        const unsigned short* Qb = Qh + (((size_t)b * H_ + h) * S_) * D_;
        const unsigned short* Kb = Kh + (((size_t)b * H_ + h) * S_) * D_;
        const unsigned short* Vb = Vt + (((size_t)b * H_ + h) * D_) * S_;

        // ---- scores: wave w owns 256 key positions; write bf16 raw scores
        v16bf aq0 = frag2(Qb + (size_t)(q0 + col16) * D_ + hi * 8,
                          Qb + (size_t)(q0 + col16) * D_ + 16 + hi * 8);
        v16bf aq1 = frag2(Qb + (size_t)(q0 + col16) * D_ + 32 + hi * 8,
                          Qb + (size_t)(q0 + col16) * D_ + 48 + hi * 8);
        for (int kt = 0; kt < 16; ++kt) {
            int kbase = wave * 256 + kt * 16;
            int kpos  = kbase + col16;
            v16bf b0 = frag2(Kb + (size_t)kpos * D_ + hi * 8,
                             Kb + (size_t)kpos * D_ + 16 + hi * 8);
            v16bf b1 = frag2(Kb + (size_t)kpos * D_ + 32 + hi * 8,
                             Kb + (size_t)kpos * D_ + 48 + hi * 8);
            v8f c = {};
            c = WMMA_BF16(aq0, b0, c);
            c = WMMA_BF16(aq1, b1, c);
            #pragma unroll
            for (int i = 0; i < 8; ++i)
                pbf[(i + hi * 8) * 2048 + kbase + col16] =
                    f2bfbits(c[i] * 0.125f);                 // 1/sqrt(64)
        }
        __syncthreads();

        // ---- softmax: 16 threads/row, blocked 128-element ranges, vector IO
        {
            int row = tid >> 4;
            int sub = tid & 15;
            unsigned short* pr = pbf + row * 2048 + sub * 128;

            float mx = -3.0e38f;
            for (int j0 = 0; j0 < 128; j0 += 8) {
                v8us v;
                __builtin_memcpy(&v, pr + j0, 16);
                #pragma unroll
                for (int t = 0; t < 8; ++t) mx = fmaxf(mx, bf2f(v[t]));
            }
            red[row * 16 + sub] = mx;
            __syncthreads();
            if (sub == 0) {
                float m2 = red[row * 16];
                for (int t = 1; t < 16; ++t) m2 = fmaxf(m2, red[row * 16 + t]);
                red[row * 16] = m2;
            }
            __syncthreads();
            float rowmax = red[row * 16];
            __syncthreads();

            float sum = 0.0f;
            for (int j0 = 0; j0 < 128; j0 += 8) {
                v8us v;
                unsigned o[4];
                __builtin_memcpy(&v, pr + j0, 16);
                float e[8];
                #pragma unroll
                for (int t = 0; t < 8; ++t) {
                    e[t] = __expf(bf2f(v[t]) - rowmax);
                    sum += e[t];
                }
                #pragma unroll
                for (int t = 0; t < 4; ++t)
                    o[t] = f2bf2bits(e[2 * t], e[2 * t + 1]);
                __builtin_memcpy(pr + j0, o, 16);       // in-place: own elements
            }
            red[row * 16 + sub] = sum;
            __syncthreads();
            if (sub == 0) {
                float s2 = 0.0f;
                for (int t = 0; t < 16; ++t) s2 += red[row * 16 + t];
                invs[row] = 1.0f / s2;
            }
            __syncthreads();

            // avg += p * inv / H   (probs stay un-normalized in pbf)
            float c16 = invs[row] * (1.0f / (float)H_);
            float* av = avg + row * 2048 + sub * 128;
            for (int j0 = 0; j0 < 128; j0 += 8) {
                v8us v;
                v8f  a;
                __builtin_memcpy(&v, pr + j0, 16);
                __builtin_memcpy(&a, av + j0, 32);
                #pragma unroll
                for (int t = 0; t < 8; ++t) a[t] += bf2f(v[t]) * c16;
                __builtin_memcpy(av + j0, &a, 32);
            }
        }
        __syncthreads();

        // ---- P @ V: 8 waves = 4 d-tiles x 2 K-halves; A from pbf, B from Vt
        {
            const int ntile = wave & 3;
            const int kh    = wave >> 2;
            v8f c = {};
            const unsigned short* prow = pbf + col16 * 2048;
            const unsigned short* vcol = Vb + (size_t)(ntile * 16 + col16) * S_;
            for (int kb = kh * 1024; kb < kh * 1024 + 1024; kb += 32) {
                v16bf a  = frag2(prow + kb + hi * 8, prow + kb + 16 + hi * 8);
                v16bf bf = frag2(vcol + kb + hi * 8, vcol + kb + 16 + hi * 8);
                c = WMMA_BF16(a, bf, c);
            }
            if (kh == 1) {
                #pragma unroll
                for (int i = 0; i < 8; ++i)
                    pvred[ntile * 256 + (i + hi * 8) * 16 + col16] = c[i];
            }
            __syncthreads();
            if (kh == 0) {
                #pragma unroll
                for (int i = 0; i < 8; ++i) {
                    int row = i + hi * 8;
                    float tot = (c[i] + pvred[ntile * 256 + row * 16 + col16])
                              * invs[row];
                    int q = q0 + row;
                    AO[((size_t)b * S_ + q) * E_ + h * D_ + ntile * 16 + col16] =
                        f2bfbits(tot);
                }
            }
        }
        __syncthreads();   // pbf / pvred reused next head
    }

    // ---- attn average [B, Sq, Sk] fp32, vectorized
    for (int e = tid * 4; e < 16 * 2048; e += 1024) {
        int r = e >> 11, cidx = e & 2047;
        *reinterpret_cast<float4*>(
            attn_avg + ((size_t)b * S_ + q0 + r) * S_ + cidx) =
            *reinterpret_cast<float4*>(avg + e);
    }
}

// ---------------------------------------------------------------------------
// Kernel 3: output projection: out = AO(bf16) @ Wo^T + bo  (fp32 out)
// ---------------------------------------------------------------------------
__global__ void __launch_bounds__(256) oproj_kernel(
    const unsigned short* __restrict__ AO, const float* __restrict__ Wo,
    const float* __restrict__ bo, float* __restrict__ out)
{
    const int tileM = blockIdx.x * 128;
    const int tileN = blockIdx.y * 64;
    const int tid   = threadIdx.x;
    const int wave  = tid >> 5;
    const int lane  = tid & 31;
    const int hi    = (lane >> 4) & 1;
    const int col16 = lane & 15;
    const int rowBase = tileM + wave * 16;

    __shared__ unsigned short As[128][40];
    __shared__ unsigned short Bs[64][40];

    v8f acc[4] = {};

    for (int k0 = 0; k0 < E_; k0 += 32) {
        #pragma unroll
        for (int e = tid; e < 128 * 4; e += 256) {       // 8-element bf16 copies
            int r = e >> 2, c8 = (e & 3) * 8;
            v8us v;
            __builtin_memcpy(&v, AO + (size_t)(tileM + r) * E_ + k0 + c8, 16);
            __builtin_memcpy(&As[r][c8], &v, 16);
        }
        #pragma unroll
        for (int e = tid; e < 64 * 8; e += 256) {
            int n = e >> 3, c4 = (e & 7) * 4;
            float4 w = *reinterpret_cast<const float4*>(
                Wo + (size_t)(tileN + n) * E_ + k0 + c4);
            unsigned long long p = pack4(w.x, w.y, w.z, w.w);
            __builtin_memcpy(&Bs[n][c4], &p, 8);
        }
        if (k0 + 32 < E_)
            __builtin_prefetch(&AO[(size_t)(tileM + wave * 16) * E_ + k0 + 32], 0, 1);
        __syncthreads();

        v16bf a = frag2(&As[wave * 16 + col16][hi * 8],
                        &As[wave * 16 + col16][16 + hi * 8]);
        #pragma unroll
        for (int j = 0; j < 4; ++j) {
            v16bf bf = frag2(&Bs[j * 16 + col16][hi * 8],
                             &Bs[j * 16 + col16][16 + hi * 8]);
            acc[j] = WMMA_BF16(a, bf, acc[j]);
        }
        __syncthreads();
    }

    #pragma unroll
    for (int j = 0; j < 4; ++j) {
        int n = tileN + j * 16 + col16;
        float bb = bo[n];
        #pragma unroll
        for (int i = 0; i < 8; ++i) {
            int m = rowBase + i + hi * 8;
            out[(size_t)m * E_ + n] = acc[j][i] + bb;
        }
    }
}

// ---------------------------------------------------------------------------
extern "C" void kernel_launch(void* const* d_in, const int* in_sizes, int n_in,
                              void* d_out, int out_size, void* d_ws, size_t ws_size,
                              hipStream_t stream)
{
    const float* q_in = (const float*)d_in[0];
    const float* k_in = (const float*)d_in[1];
    const float* v_in = (const float*)d_in[2];
    const float* Wq   = (const float*)d_in[3];
    const float* bq   = (const float*)d_in[4];
    const float* Wk   = (const float*)d_in[5];
    const float* bk   = (const float*)d_in[6];
    const float* Wv   = (const float*)d_in[7];
    const float* bv   = (const float*)d_in[8];
    const float* Wo   = (const float*)d_in[9];
    const float* bo   = (const float*)d_in[10];

    const size_t nBHSD = (size_t)B_ * H_ * S_ * D_;   // 8,388,608
    unsigned short* Qh = (unsigned short*)d_ws;
    unsigned short* Kh = Qh + nBHSD;
    unsigned short* Vt = Kh + nBHSD;                  // [B,H,D,S]
    unsigned short* AO = Vt + nBHSD;

    float* out      = (float*)d_out;
    float* attn_avg = out + (size_t)B_ * S_ * E_;

    qkv_rope_kernel<<<dim3(64, 16, 3), 256, 0, stream>>>(
        q_in, k_in, v_in, Wq, bq, Wk, bk, Wv, bv, Qh, Kh, Vt);

    // pbf(16384 f32-equiv) + avg(32768) + red(256) + invs(16) + pvred(1024)
    size_t smem = (size_t)(16384 + 32768 + 256 + 16 + 1024) * sizeof(float);
    hipFuncSetAttribute(reinterpret_cast<const void*>(attn_kernel),
                        hipFuncAttributeMaxDynamicSharedMemorySize, (int)smem);
    attn_kernel<<<dim3(S_ / 16, B_), 256, smem, stream>>>(Qh, Kh, Vt, AO, attn_avg);

    oproj_kernel<<<dim3(64, 16), 256, 0, stream>>>(AO, Wo, bo, out);
}
